// NuGraphCore_52948356825594
// MI455X (gfx1250) — matrix operand order repaired and checked by reading, not compiled, not run on hardware
//
#include <hip/hip_runtime.h>
#include <hip/hip_bf16.h>
#include <math.h>

// ---------------------------------------------------------------------------
// NuGraphCore forward for MI455X (gfx1250, wave32).
//  - Dense linears / attention projections: f16 WMMA (v_wmma_f32_16x16x32_f16),
//    fp32 accumulate, vectorized fp32->f16 tile staging in LDS, b128 fragment
//    loads in contiguous k order (k = 16*hsel + e), same for A and B.
//  - Edge message passing: fused VALU kernel, gate.w staged in LDS, segment
//    softmax via ordered-uint atomicMax + f32 atomicAdd in L2.
//  - Cross-attention: fused online-softmax (no [Nq,Nk] score matrix).
// ---------------------------------------------------------------------------

typedef __attribute__((ext_vector_type(16))) _Float16 v16h;
typedef __attribute__((ext_vector_type(8)))  _Float16 h8;
typedef __attribute__((ext_vector_type(4)))  _Float16 h4;
typedef __attribute__((ext_vector_type(4)))  float    f4v;
typedef __attribute__((ext_vector_type(8)))  float    v8f;

__device__ inline float wave_sum(float v) {
#pragma unroll
  for (int o = 16; o > 0; o >>= 1) v += __shfl_xor(v, o, 32);
  return v;
}
// monotonic float <-> uint mapping for atomicMax-based segment max
__device__ inline unsigned ford(float f) {
  unsigned u = __float_as_uint(f);
  return (u & 0x80000000u) ? ~u : (u | 0x80000000u);
}
__device__ inline float funord(unsigned u) {
  u = (u & 0x80000000u) ? (u & 0x7fffffffu) : ~u;
  return __uint_as_float(u);
}
__device__ inline float sigmoidf_(float x) { return 1.f / (1.f + expf(-x)); }
__device__ inline float mishf_(float x) {
  float sp = (x > 20.f) ? x : log1pf(expf(x));
  return x * tanhf(sp);
}

// ---------------------------------------------------------------------------
__global__ void fill_u32(unsigned* __restrict__ p, unsigned v, long long n) {
  long long i = (long long)blockIdx.x * blockDim.x + threadIdx.x;
  if (i < n) p[i] = v;
}

// ---------------------------------------------------------------------------
// WMMA GEMM: C[M,N] = act(A[M,K] @ B[K,N] + bias).  A,B,bias fp32 row-major.
// Block = 256 threads = 8 waves; block tile 128x32.  Each wave owns one
// 16-row strip and two 16-col tiles -> 2 WMMA per K-step sharing one A frag.
// Requires K % 32 == 0 and N % 32 == 0 (true for every GEMM in this model).
// ---------------------------------------------------------------------------
#define GBM 128
#define GBN 32
#define GBK 32
#define LDSTR 40   // LDS row stride in halves: 80 B, 16B-aligned, non-pow2

__global__ __launch_bounds__(256)
void gemm_wmma_f16(const float* __restrict__ A, const float* __restrict__ B,
                   const float* __restrict__ bias, float* __restrict__ C,
                   int M, int N, int K, int act /*0=none,1=mish*/) {
  __shared__ _Float16 As[GBM * LDSTR];   // [row][k]  10.0 KB
  __shared__ _Float16 BsT[GBN * LDSTR];  // [col][k]   2.5 KB
  const int tid  = threadIdx.x;
  const int lane = tid & 31;
  const int wv   = tid >> 5;
  const int hsel = lane >> 4;   // half-wave select (lanes 0-15 / 16-31)
  const int rc   = lane & 15;   // row (A) / col (B,C)
  const int m0 = blockIdx.y * GBM;
  const int n0 = blockIdx.x * GBN;
  const bool fastA = (m0 + GBM) <= M;   // K,N are multiples of 32 by contract

  v8f acc0 = {}, acc1 = {};
  for (int k0 = 0; k0 < K; k0 += GBK) {
    // ---- stage A tile fp32 -> f16 (128x32) ----
    if (fastA) {
      // 1024 float4 chunks, 4 per thread: b128 global load -> b64 ds store
#pragma unroll
      for (int t = 0; t < 4; ++t) {
        int cid = tid + t * 256;
        int r = cid >> 3, c4 = (cid & 7) * 4;
        f4v f = *(const f4v*)(A + (size_t)(m0 + r) * K + k0 + c4);
        h4 h;
#pragma unroll
        for (int j = 0; j < 4; ++j) h[j] = (_Float16)f[j];
        *(h4*)&As[r * LDSTR + c4] = h;
      }
    } else {
      for (int i = tid; i < GBM * GBK; i += 256) {
        int r = i >> 5, kk = i & 31;
        int gm = m0 + r;
        float v = (gm < M) ? A[(size_t)gm * K + k0 + kk] : 0.f;
        As[r * LDSTR + kk] = (_Float16)v;
      }
    }
    // ---- stage B tile fp32 -> f16, transposed to [col][k] (32x32) ----
    {
      int kk = tid >> 3, c4 = (tid & 7) * 4;
      f4v f = *(const f4v*)(B + (size_t)(k0 + kk) * N + n0 + c4);
#pragma unroll
      for (int j = 0; j < 4; ++j) BsT[(c4 + j) * LDSTR + kk] = (_Float16)f[j];
    }
    // prefetch next A tile (lowers to global_prefetch_b8)
    if (fastA && (k0 + GBK < K)) {
      int r = tid >> 1;
      __builtin_prefetch(&A[(size_t)(m0 + r) * K + k0 + GBK], 0, 1);
    }
    __syncthreads();

    // ---- fragments (contiguous k = 16*hsel + e, same for A and B): b128 LDS reads ----
    const _Float16* ra = &As[(wv * 16 + rc) * LDSTR];
    h8 a0 = *(const h8*)(ra + 16 * hsel);       // e=0..7 : k = 16*hsel + e
    h8 a1 = *(const h8*)(ra + 16 * hsel + 8);   // e=8..15: k = 16*hsel + e
    v16h af;
#pragma unroll
    for (int e = 0; e < 8; ++e) { af[e] = a0[e]; af[e + 8] = a1[e]; }

    const _Float16* rb0 = &BsT[rc * LDSTR];          // tile 0: col = n0 + rc
    const _Float16* rb1 = &BsT[(16 + rc) * LDSTR];   // tile 1: col = n0 + 16 + rc
    h8 b00 = *(const h8*)(rb0 + 16 * hsel);
    h8 b01 = *(const h8*)(rb0 + 16 * hsel + 8);
    h8 b10 = *(const h8*)(rb1 + 16 * hsel);
    h8 b11 = *(const h8*)(rb1 + 16 * hsel + 8);
    v16h bf0, bf1;
#pragma unroll
    for (int e = 0; e < 8; ++e) {
      bf0[e] = b00[e]; bf0[e + 8] = b01[e];
      bf1[e] = b10[e]; bf1[e + 8] = b11[e];
    }
    acc0 = __builtin_amdgcn_wmma_f32_16x16x32_f16(false, af, false, bf0,
                                                  (short)0, acc0, false, false);
    acc1 = __builtin_amdgcn_wmma_f32_16x16x32_f16(false, af, false, bf1,
                                                  (short)0, acc1, false, false);
    __syncthreads();
  }

  // ---- epilogue: C/D layout VGPR r -> row r + 8*hsel, col = rc ----
#pragma unroll
  for (int r = 0; r < 8; ++r) {
    int row = m0 + wv * 16 + r + 8 * hsel;
    if (row < M) {
      int c0 = n0 + rc, c1 = n0 + 16 + rc;
      float x0 = acc0[r] + (bias ? bias[c0] : 0.f);
      float x1 = acc1[r] + (bias ? bias[c1] : 0.f);
      if (act == 1) { x0 = mishf_(x0); x1 = mishf_(x1); }
      C[(size_t)row * N + c0] = x0;
      C[(size_t)row * N + c1] = x1;
    }
  }
}

// ---------------------------------------------------------------------------
// Fused edge kernel (one wave per edge, 8 waves/block).  Computes
//   edge_w = sigmoid(We . [x_i, x_j]); new_msg = edge_w * x_j
//   gate   = sigmoid([new_msg, x_j] @ Wg + bg); msg = gate*nm + (1-gate)*x_j
// pass 0: atomicMax of ordered(msg) into seg_m
// pass 1: ex = exp(msg - max); atomicAdd seg_den += ex, seg_num += ex*msg
// ---------------------------------------------------------------------------
template <int F>
__global__ __launch_bounds__(256)
void edge_pass(const float* __restrict__ xs, const float* __restrict__ xd,
               const int* __restrict__ src, const int* __restrict__ dst,
               long long E,
               const float* __restrict__ We, const float* __restrict__ be,
               const float* __restrict__ Wg, const float* __restrict__ bg,
               unsigned* __restrict__ seg_m, float* __restrict__ seg_den,
               float* __restrict__ seg_num, int pass) {
  __shared__ float sWg[2 * F * F];
  __shared__ float sWe[2 * F];
  __shared__ float sNm[8][F];
  __shared__ float sXj[8][F];
  for (int i = threadIdx.x; i < 2 * F * F; i += 256) sWg[i] = Wg[i];
  for (int i = threadIdx.x; i < 2 * F; i += 256) sWe[i] = We[i];
  __syncthreads();

  const int lane = threadIdx.x & 31;
  const int wv   = threadIdx.x >> 5;
  const float eb = be[0];

  for (long long e = (long long)blockIdx.x * 8 + wv; e < E;
       e += (long long)gridDim.x * 8) {
    int s = src[e], d = dst[e];
    const float* xj = xs + (size_t)s * F;
    const float* xi = xd + (size_t)d * F;

    float a = 0.f;
    for (int i = lane; i < 2 * F; i += 32)
      a += ((i < F) ? xi[i] : xj[i - F]) * sWe[i];
    float ew = sigmoidf_(wave_sum(a) + eb);

    for (int f = lane; f < F; f += 32) {
      float x = xj[f];
      sXj[wv][f] = x;
      sNm[wv][f] = ew * x;
    }
    // wave-synchronous LDS exchange: ensure DS writes landed before reads
    asm volatile("s_wait_dscnt 0" ::: "memory");

    for (int f = lane; f < F; f += 32) {
      float g = bg[f];
      for (int k = 0; k < 2 * F; ++k) {
        float in = (k < F) ? sNm[wv][k] : sXj[wv][k - F];
        g = fmaf(in, sWg[k * F + f], g);
      }
      g = sigmoidf_(g);
      float nm = sNm[wv][f], xv = sXj[wv][f];
      float msg = g * nm + (1.f - g) * xv;
      size_t idx = (size_t)d * F + f;
      if (pass == 0) {
        atomicMax(&seg_m[idx], ford(msg));
      } else {
        float ex = expf(msg - funord(seg_m[idx]));
        atomicAdd(&seg_den[idx], ex);
        atomicAdd(&seg_num[idx], ex * msg);
      }
    }
  }
}

// ---------------------------------------------------------------------------
// aggr = num/den (0 for empty segments); cat = [aggr, x_dst]  -> [n, 2F]
// ---------------------------------------------------------------------------
__global__ void finish_cat(const float* __restrict__ num,
                           const float* __restrict__ den,
                           const float* __restrict__ xd, float* __restrict__ cat,
                           long long n, int F) {
  long long i = (long long)blockIdx.x * blockDim.x + threadIdx.x;
  if (i >= n * F) return;
  long long r = i / F;
  int f = (int)(i % F);
  float dn = den[i];
  float a = dn > 0.f ? num[i] / dn : 0.f;
  cat[r * 2 * F + f]     = a;
  cat[r * 2 * F + F + f] = xd[i];
}

// ---------------------------------------------------------------------------
// y = LayerNorm(a + b) * g + be  (one wave per row, D in {32,64})
// ---------------------------------------------------------------------------
__global__ __launch_bounds__(256)
void add_ln(const float* __restrict__ a, const float* __restrict__ b,
            const float* __restrict__ g, const float* __restrict__ be,
            float* __restrict__ y, int n, int D) {
  const int lane = threadIdx.x & 31;
  const int wv   = threadIdx.x >> 5;
  const int row  = blockIdx.x * 8 + wv;
  if (row >= n) return;
  const int cnt = D >> 5;  // 1 or 2
  float v[2];
  float s = 0.f, s2 = 0.f;
  for (int t = 0; t < cnt; ++t) {
    float x = a[(size_t)row * D + t * 32 + lane] +
              b[(size_t)row * D + t * 32 + lane];
    v[t] = x; s += x; s2 += x * x;
  }
  s  = wave_sum(s);
  s2 = wave_sum(s2);
  float mu  = s / D;
  float inv = rsqrtf(s2 / D - mu * mu + 1e-5f);
  for (int t = 0; t < cnt; ++t) {
    int f = t * 32 + lane;
    y[(size_t)row * D + f] = (v[t] - mu) * inv * g[f] + be[f];
  }
}

// ---------------------------------------------------------------------------
// Fused attention: O[Nq,256] = softmax(Q K^T / 8) V, online softmax.
// One wave per query row; 16-key K/V tiles staged in LDS (32 KB).
// ---------------------------------------------------------------------------
__global__ __launch_bounds__(256)
void attn_fused(const float* __restrict__ Q, const float* __restrict__ Km,
                const float* __restrict__ Vm, float* __restrict__ O,
                int Nq, int Nk) {
  __shared__ float sK[16][256];
  __shared__ float sV[16][256];
  const int lane = threadIdx.x & 31;
  const int wv   = threadIdx.x >> 5;
  const int row  = blockIdx.x * 8 + wv;
  const bool active = row < Nq;

  float q[8], o[8];
  float m = -3.0e38f, l = 0.f;
#pragma unroll
  for (int t = 0; t < 8; ++t) {
    o[t] = 0.f;
    q[t] = active ? Q[(size_t)row * 256 + t * 32 + lane] : 0.f;
  }

  for (int j0 = 0; j0 < Nk; j0 += 16) {
    int jn = (Nk - j0 < 16) ? (Nk - j0) : 16;
    __syncthreads();
    for (int i = threadIdx.x; i < jn * 256; i += 256) {
      int jj = i >> 8, c = i & 255;
      sK[jj][c] = Km[(size_t)(j0 + jj) * 256 + c];
      sV[jj][c] = Vm[(size_t)(j0 + jj) * 256 + c];
    }
    __syncthreads();
    if (active) {
      for (int jj = 0; jj < jn; ++jj) {
        float p = 0.f;
#pragma unroll
        for (int t = 0; t < 8; ++t) p += q[t] * sK[jj][t * 32 + lane];
        p = wave_sum(p) * 0.125f;  // 1/sqrt(HEAD_DIM=64)
        float mn = fmaxf(m, p);
        float cf = expf(m - mn);
        float w  = expf(p - mn);
        l = l * cf + w;
#pragma unroll
        for (int t = 0; t < 8; ++t) o[t] = o[t] * cf + w * sV[jj][t * 32 + lane];
        m = mn;
      }
    }
  }
  if (active) {
    float invl = 1.f / l;
#pragma unroll
    for (int t = 0; t < 8; ++t)
      O[(size_t)row * 256 + t * 32 + lane] = o[t] * invl;
  }
}

// ===========================================================================
// Host orchestration
// ===========================================================================
static inline int idiv_(int a, int b) { return (a + b - 1) / b; }

extern "C" void kernel_launch(void* const* d_in, const int* in_sizes, int n_in,
                              void* d_out, int out_size, void* d_ws,
                              size_t ws_size, hipStream_t stream) {
  (void)in_sizes; (void)n_in; (void)out_size; (void)ws_size;

  const int N_HIT = 100000, N_SP = 25000, N_EVT = 16, N_PMT = 5000, N_FLASH = 64;
  const int HID = 256;
  const long long E_HH = 800000, E_HS = 200000, E_SE = 25000, E_OP = 40000;

  auto FP = [&](int i) { return (const float*)d_in[i]; };
  auto IP = [&](int i) { return (const int*)d_in[i]; };

  const float* hit_x   = FP(0);
  const float* sp_x    = FP(1);
  const float* evt_x   = FP(2);
  const float* ophit_x = FP(3);
  const float* pmt_x   = FP(4);
  const float* flash_x = FP(5);
  const int* ehh     = IP(6);
  const int* hh_src  = ehh;
  const int* hh_dst  = ehh + E_HH;
  const int* hs_src  = IP(7);
  const int* hs_dst  = IP(8);
  const int* se_src  = IP(9);
  const int* se_dst  = IP(10);
  const int* op_src  = IP(11);
  const int* op_dst  = IP(12);
  // params flattened in setup_inputs() insertion order:
  // block: edge.w, edge.b, gate.w, gate.b, net1.w, net1.b, net2.w, net2.b
  // xattn: q.w,q.b, k.w,k.b, v.w,v.b, attn.w,attn.b, out.w,out.b, norm.g,norm.b
  const int PB_PLANE = 13, PB_P2N = 21, PB_N2I = 29, PB_OP = 37;
  const int PB_HFA = 45, PB_NPA = 57;
  const int HIT_NORM = 69, SP_NORM = 71, EVT_NORM = 73, PMT_NORM = 75;

  // ---- workspace arena (stage-wise overlays; peak ~241 MB) ----
  char* ws = (char*)d_ws;
  float*    segDen = (float*)(ws + 0);           // 25.6 MB
  float*    segNum = (float*)(ws + 25600000);    // 25.6 MB
  unsigned* segM   = (unsigned*)(ws + 51200000); // 25.6 MB
  float*    catB   = (float*)(ws + 76800000);    // 51.2 MB
  float*    hB     = (float*)(ws + 128000000);   // 25.6 MB
  float*    blkB   = (float*)(ws + 153600000);   // 25.6 MB
  // attention overlay (block-stage scratch is dead by the time these run)
  float* Qb = (float*)(ws + 0);                  // 102.4 MB
  float* Kb = (float*)(ws + 102400000);          // 5.12 MB
  float* Vb = (float*)(ws + 107520000);          // 5.12 MB
  float* Ob = (float*)(ws + 112640000);          // 102.4 MB
  float* T1 = Qb;                                // reuse (Q dead after attn_fused)
  float* T2 = Ob;                                // reuse (O dead after attn gemm)
  float* curB = (float*)(ws + 215040000);        // 25.6 MB, survives attention

  float* outHit = (float*)d_out;
  float* outSp  = outHit + (size_t)N_HIT * 64;
  float* outEvt = outSp + (size_t)N_SP * 64;
  float* outPmt = outEvt + (size_t)N_EVT * 64;

  auto gemm = [&](const float* A, const float* B, const float* bias, float* C,
                  int M, int N, int K, int act) {
    gemm_wmma_f16<<<dim3(idiv_(N, GBN), idiv_(M, GBM)), 256, 0, stream>>>(
        A, B, bias, C, M, N, K, act);
  };

  auto run_block = [&](int F, const float* xs, const float* xd, const int* src,
                       const int* dst, int n_dst, long long E, int pb,
                       float* out) {
    long long segN = (long long)n_dst * F;
    unsigned fg = (unsigned)((segN + 255) / 256);
    fill_u32<<<fg, 256, 0, stream>>>(segM, 0u, segN);
    fill_u32<<<fg, 256, 0, stream>>>((unsigned*)segDen, 0u, segN);
    fill_u32<<<fg, 256, 0, stream>>>((unsigned*)segNum, 0u, segN);
    long long eb = (E + 7) / 8;
    unsigned eg = (unsigned)(eb > 4096 ? 4096 : eb);
    if (F == 64) {
      edge_pass<64><<<eg, 256, 0, stream>>>(xs, xd, src, dst, E, FP(pb + 0),
          FP(pb + 1), FP(pb + 2), FP(pb + 3), segM, segDen, segNum, 0);
      edge_pass<64><<<eg, 256, 0, stream>>>(xs, xd, src, dst, E, FP(pb + 0),
          FP(pb + 1), FP(pb + 2), FP(pb + 3), segM, segDen, segNum, 1);
    } else {
      edge_pass<32><<<eg, 256, 0, stream>>>(xs, xd, src, dst, E, FP(pb + 0),
          FP(pb + 1), FP(pb + 2), FP(pb + 3), segM, segDen, segNum, 0);
      edge_pass<32><<<eg, 256, 0, stream>>>(xs, xd, src, dst, E, FP(pb + 0),
          FP(pb + 1), FP(pb + 2), FP(pb + 3), segM, segDen, segNum, 1);
    }
    finish_cat<<<fg, 256, 0, stream>>>(segNum, segDen, xd, catB, n_dst, F);
    gemm(catB, FP(pb + 4), FP(pb + 5), hB, n_dst, F, 2 * F, 1);   // net1+mish
    gemm(hB, FP(pb + 6), FP(pb + 7), out, n_dst, F, F, 1);        // net2+mish
  };

  auto run_xattn = [&](int qd, int kd, int Nq, int Nk, const float* x,
                       const float* ctx, int pb, float* y) {
    gemm(x,   FP(pb + 0), FP(pb + 1), Qb, Nq, HID, qd, 0);
    gemm(ctx, FP(pb + 2), FP(pb + 3), Kb, Nk, HID, kd, 0);
    gemm(ctx, FP(pb + 4), FP(pb + 5), Vb, Nk, HID, kd, 0);
    attn_fused<<<idiv_(Nq, 8), 256, 0, stream>>>(Qb, Kb, Vb, Ob, Nq, Nk);
    gemm(Ob, FP(pb + 6), FP(pb + 7), T1, Nq, HID, HID, 0);
    gemm(T1, FP(pb + 8), FP(pb + 9), T2, Nq, qd, HID, 0);
    add_ln<<<idiv_(Nq, 8), 256, 0, stream>>>(T2, x, FP(pb + 10), FP(pb + 11),
                                             y, Nq, qd);
  };

  // ---- 1. plane_net: hit <- hit, then hit/flash cross-attention ----
  run_block(64, hit_x, hit_x, hh_src, hh_dst, N_HIT, E_HH, PB_PLANE, blkB);
  add_ln<<<idiv_(N_HIT, 8), 256, 0, stream>>>(blkB, hit_x, FP(HIT_NORM),
                                              FP(HIT_NORM + 1), curB, N_HIT, 64);
  run_xattn(64, 32, N_HIT, N_FLASH, curB, flash_x, PB_HFA, outHit);

  // ---- 2. plane_to_nexus: hit -> sp, then sp/pmt cross-attention ----
  run_block(64, outHit, sp_x, hs_src, hs_dst, N_SP, E_HS, PB_P2N, blkB);
  add_ln<<<idiv_(N_SP, 8), 256, 0, stream>>>(blkB, sp_x, FP(SP_NORM),
                                             FP(SP_NORM + 1), curB, N_SP, 64);
  run_xattn(64, 32, N_SP, N_PMT, curB, pmt_x, PB_NPA, outSp);

  // ---- 3. nexus_to_interaction: sp -> evt ----
  run_block(64, outSp, evt_x, se_src, se_dst, N_EVT, E_SE, PB_N2I, blkB);
  add_ln<<<idiv_(N_EVT, 8), 256, 0, stream>>>(blkB, evt_x, FP(EVT_NORM),
                                              FP(EVT_NORM + 1), outEvt, N_EVT, 64);

  // ---- 4. ophit_to_pmt: ophit -> pmt ----
  run_block(32, ophit_x, pmt_x, op_src, op_dst, N_PMT, E_OP, PB_OP, blkB);
  add_ln<<<idiv_(N_PMT, 8), 256, 0, stream>>>(blkB, pmt_x, FP(PMT_NORM),
                                              FP(PMT_NORM + 1), outPmt, N_PMT, 32);
}
